// GATv2_15556371546778
// MI455X (gfx1250) — matrix operand order repaired and checked by reading, not compiled, not run on hardware
//
#include <hip/hip_runtime.h>
#include <hip/hip_bf16.h>
#include <cstdint>

typedef __attribute__((ext_vector_type(16))) _Float16 v16h;
typedef __attribute__((ext_vector_type(8)))  float    v8f;

#define NEG_SLOPE 0.2f
#define USE_ASYNC_STAGE 1

// ---------------------------------------------------------------------------
// Precision prep: activations f32->f16, weights f32 [I x O] -> f16 [O x I]^T.
// Transposed f16 weights make a lane's B fragment (16 consecutive K values of
// one output column) a single contiguous 32-byte load.
// ---------------------------------------------------------------------------
__global__ void cvt_x_f16(const float* __restrict__ x, _Float16* __restrict__ xh, int n) {
    int i = blockIdx.x * blockDim.x + threadIdx.x;
    if (i < n) xh[i] = (_Float16)x[i];
}

__global__ void cvt_w_f16t(const float* __restrict__ W, _Float16* __restrict__ Wt,
                           int I, int O) {
    int i = blockIdx.x * blockDim.x + threadIdx.x;
    if (i >= I * O) return;
    int r = i / O, c = i - r * O;            // W[r][c]
    Wt[(size_t)c * I + r] = (_Float16)W[i];  // Wt[c][r]
}

// ---------------------------------------------------------------------------
// WMMA GEMM: Y[N x O] = Xh[N x I] @ Wt[O x I]^T, f16 mul / f32 acc.
// block = 256 threads (8 wave32) covering 32 rows x 128 cols of Y.
// Each wave: one N-tile (16 cols), two M-tiles sharing each B fragment
// -> 2 WMMAs per B load, I/32 * 2 straight-line v_wmma per kernel.
// grid = (ceil(N/32), O/128). Requires O%128==0, I%32==0.
// ---------------------------------------------------------------------------
template<int I>
__global__ __launch_bounds__(256)
void gatv2_gemm_wmma(const _Float16* __restrict__ Xh,
                     const _Float16* __restrict__ Wt,
                     float* __restrict__ Y,
                     int Nrows, int O) {
    __shared__ _Float16 sX[32 * I];

    const int tid  = threadIdx.x;
    const int row0 = blockIdx.x * 32;
    const int col0 = blockIdx.y * 128;

    // Stage 32 x I activation rows into LDS, 16 bytes per thread-iteration.
    // Row index clamped for the final partial tile (junk rows, stores guarded).
    const int nchunk = (32 * I) / 8;         // 8 halves per 16B chunk
    for (int c = tid; c < nchunk; c += 256) {
        int idx = c * 8;
        int r   = idx / I;
        int col = idx - r * I;
        int row = row0 + r;
        if (row >= Nrows) row = Nrows - 1;
        const uint4* gsrc = (const uint4*)(Xh + (size_t)row * I + col);
#if USE_ASYNC_STAGE
        unsigned lds = (unsigned)(uintptr_t)(&sX[idx]);   // low 32b = LDS offset
        unsigned long long ga = (unsigned long long)(uintptr_t)gsrc;
        asm volatile("global_load_async_to_lds_b128 %0, %1, off"
                     :: "v"(lds), "v"(ga) : "memory");
#else
        *(uint4*)(&sX[idx]) = *gsrc;
#endif
    }
#if USE_ASYNC_STAGE
    asm volatile("s_wait_asynccnt 0" ::: "memory");
#endif
    __syncthreads();

    const int wave = tid >> 5;
    const int lane = tid & 31;
    const int n    = col0 + wave * 16 + (lane & 15);
    const int mrow = lane & 15;
    const int kbA  = (lane < 16) ? 0 : 16;   // A: K striped across lane halves (matches B)
    const int kbB  = (lane < 16) ? 0 : 16;   // B: K striped across lane halves

    const _Float16* wrow = Wt + (size_t)n * I;

    v8f acc0 = {}, acc1 = {};
#pragma unroll
    for (int k0 = 0; k0 < I; k0 += 32) {
        v16h b = *(const v16h*)(wrow + k0 + kbB);   // 16 consecutive K, one load
        v16h a0, a1;
#pragma unroll
        for (int e = 0; e < 8; ++e) {
            a0[e]     = sX[mrow * I + k0 + kbA + e];
            a0[e + 8] = sX[mrow * I + k0 + kbA + 8 + e];
            a1[e]     = sX[(16 + mrow) * I + k0 + kbA + e];
            a1[e + 8] = sX[(16 + mrow) * I + k0 + kbA + 8 + e];
        }
        acc0 = __builtin_amdgcn_wmma_f32_16x16x32_f16(
            false, a0, false, b, (short)0, acc0, false, false);
        acc1 = __builtin_amdgcn_wmma_f32_16x16x32_f16(
            false, a1, false, b, (short)0, acc1, false, false);
    }

    // C/D layout: VGPR r -> M = r (lanes 0-15) or r+8 (lanes 16-31), N = lane%16
    const int mo = (lane < 16) ? 0 : 8;
#pragma unroll
    for (int r = 0; r < 8; ++r) {
        int rowA = row0 + mo + r;
        int rowB = row0 + 16 + mo + r;
        if (rowA < Nrows) Y[(size_t)rowA * O + n] = acc0[r];
        if (rowB < Nrows) Y[(size_t)rowB * O + n] = acc1[r];
    }
}

// ---------------------------------------------------------------------------
// Per-layer state init: nmax (mapped -inf == 0u), nsum = 0, accumulator = 0
// ---------------------------------------------------------------------------
__global__ void gatv2_init(unsigned* __restrict__ nmax, float* __restrict__ nsum,
                           float* __restrict__ acc, int N, int NO) {
    int i = blockIdx.x * blockDim.x + threadIdx.x;
    if (i < N) { nmax[i] = 0u; nsum[i] = 0.0f; }
    if (i < NO) acc[i] = 0.0f;
}

__device__ __forceinline__ unsigned f32_to_ord(float f) {
    unsigned u = __float_as_uint(f);
    return (u & 0x80000000u) ? ~u : (u | 0x80000000u);
}
__device__ __forceinline__ float ord_to_f32(unsigned u) {
    return (u & 0x80000000u) ? __uint_as_float(u & 0x7FFFFFFFu)
                             : __uint_as_float(~u);
}

// ---------------------------------------------------------------------------
// Edge scores: e = att . leaky_relu(xl[src] + xr[dst]); segment max via
// ordered-uint atomicMax. One wave32 per edge; xl/xr are L2-resident.
// ---------------------------------------------------------------------------
__global__ __launch_bounds__(256)
void gatv2_edge_score(const float* __restrict__ xl, const float* __restrict__ xr,
                      const float* __restrict__ att,
                      const int* __restrict__ src, const int* __restrict__ dst,
                      float* __restrict__ score, unsigned* __restrict__ nmax,
                      int E, int O) {
    int e = blockIdx.x * 8 + (threadIdx.x >> 5);
    if (e >= E) return;
    int lane = threadIdx.x & 31;
    size_t s = (size_t)src[e] * O, d = (size_t)dst[e] * O;
    __builtin_prefetch(&xl[s], 0, 0);
    __builtin_prefetch(&xr[d], 0, 0);

    float acc = 0.0f;
    for (int o = lane; o < O; o += 32) {
        float v = xl[s + o] + xr[d + o];
        v = (v > 0.0f) ? v : v * NEG_SLOPE;
        acc += v * att[o];
    }
#pragma unroll
    for (int off = 16; off > 0; off >>= 1)
        acc += __shfl_down(acc, off, 32);

    if (lane == 0) {
        score[e] = acc;
        atomicMax(&nmax[dst[e]], f32_to_ord(acc));
    }
}

// ---------------------------------------------------------------------------
// w = exp(score - max[dst]); segment sum via GLOBAL_ATOMIC_ADD_F32
// ---------------------------------------------------------------------------
__global__ void gatv2_edge_expsum(const float* __restrict__ score,
                                  const int* __restrict__ dst,
                                  const unsigned* __restrict__ nmax,
                                  float* __restrict__ wexp,
                                  float* __restrict__ nsum, int E) {
    int e = blockIdx.x * blockDim.x + threadIdx.x;
    if (e >= E) return;
    int d = dst[e];
    float w = __expf(score[e] - ord_to_f32(nmax[d]));
    wexp[e] = w;
    atomicAdd(&nsum[d], w);
}

// ---------------------------------------------------------------------------
// out[dst] += alpha * xl[src].  One wave32 per edge; per-lane f32 atomics.
// ---------------------------------------------------------------------------
__global__ __launch_bounds__(256)
void gatv2_edge_aggregate(const float* __restrict__ xl,
                          const float* __restrict__ wexp,
                          const float* __restrict__ nsum,
                          const int* __restrict__ src, const int* __restrict__ dst,
                          float* __restrict__ out, int E, int O) {
    int e = blockIdx.x * 8 + (threadIdx.x >> 5);
    if (e >= E) return;
    int lane = threadIdx.x & 31;
    int d = dst[e];
    float alpha = wexp[e] / (nsum[d] + 1e-16f);
    size_t s = (size_t)src[e] * O;
    __builtin_prefetch(&xl[s], 0, 0);
    size_t o0 = (size_t)d * O;
    for (int o = lane; o < O; o += 32)
        atomicAdd(&out[o0 + o], alpha * xl[s + o]);
}

// ---------------------------------------------------------------------------
// x = relu(acc + b)  (+ sigmoid on the final layer)
// ---------------------------------------------------------------------------
__global__ void gatv2_finalize(const float* __restrict__ acc,
                               const float* __restrict__ b,
                               float* __restrict__ xout,
                               int NO, int O, int last) {
    int i = blockIdx.x * blockDim.x + threadIdx.x;
    if (i >= NO) return;
    float v = acc[i] + b[i % O];
    v = fmaxf(v, 0.0f);
    if (last) v = 1.0f / (1.0f + __expf(-v));
    xout[i] = v;
}

// ---------------------------------------------------------------------------
extern "C" void kernel_launch(void* const* d_in, const int* in_sizes, int n_in,
                              void* d_out, int out_size, void* d_ws, size_t ws_size,
                              hipStream_t stream) {
    const float* x_in = (const float*)d_in[0];
    const int*   ei   = (const int*)d_in[1];
    const int E = in_sizes[1] / 2;          // 650000
    const int N = in_sizes[0] / 128;        // 10000
    const int* src = ei;
    const int* dst = ei + E;

    // workspace partition (float-sized slots; all offsets 32B-aligned)
    float* ws = (float*)d_ws;
    const size_t NH = (size_t)N * 256;
    float*     xl    = ws;                       // N*256 f32
    float*     xr    = xl + NH;                  // N*256 f32
    float*     xa    = xr + NH;                  // N*256 f32  ping
    float*     xb    = xa + NH;                  // N*256 f32  pong
    float*     score = xb + NH;                  // E f32
    float*     wexp  = score + E;                // E f32
    float*     nsum  = wexp + E;                 // N f32
    unsigned*  nmax  = (unsigned*)(nsum + N);    // N u32
    _Float16*  xh    = (_Float16*)(nmax + N);    // N*256 f16
    _Float16*  Wt    = (_Float16*)((float*)xh + NH / 2);  // 256*256 f16

    struct LayerP { const float *Wl, *Wr, *att, *b; int I, O; };
    LayerP L[3] = {
        {(const float*)d_in[2],  (const float*)d_in[3],  (const float*)d_in[4],  (const float*)d_in[5],  128, 256},
        {(const float*)d_in[6],  (const float*)d_in[7],  (const float*)d_in[8],  (const float*)d_in[9],  256, 256},
        {(const float*)d_in[10], (const float*)d_in[11], (const float*)d_in[12], (const float*)d_in[13], 256, 128},
    };

    const float* xin     = x_in;
    float*       accs[3] = {xa, xb, xa};
    float*       outs[3] = {xa, xb, (float*)d_out};

    for (int li = 0; li < 3; ++li) {
        const int I = L[li].I, O = L[li].O;
        const int NO = N * O;
        const int IO = I * O;
        float* acc = accs[li];

        // f16 prep
        cvt_x_f16<<<(N * I + 255) / 256, 256, 0, stream>>>(xin, xh, N * I);

        dim3 ggrid((N + 31) / 32, O / 128);
        cvt_w_f16t<<<(IO + 255) / 256, 256, 0, stream>>>(L[li].Wl, Wt, I, O);
        if (I == 128) gatv2_gemm_wmma<128><<<ggrid, 256, 0, stream>>>(xh, Wt, xl, N, O);
        else          gatv2_gemm_wmma<256><<<ggrid, 256, 0, stream>>>(xh, Wt, xl, N, O);

        cvt_w_f16t<<<(IO + 255) / 256, 256, 0, stream>>>(L[li].Wr, Wt, I, O);
        if (I == 128) gatv2_gemm_wmma<128><<<ggrid, 256, 0, stream>>>(xh, Wt, xr, N, O);
        else          gatv2_gemm_wmma<256><<<ggrid, 256, 0, stream>>>(xh, Wt, xr, N, O);

        gatv2_init<<<(NO + 255) / 256, 256, 0, stream>>>(nmax, nsum, acc, N, NO);

        gatv2_edge_score<<<(E + 7) / 8, 256, 0, stream>>>(
            xl, xr, L[li].att, src, dst, score, nmax, E, O);

        gatv2_edge_expsum<<<(E + 255) / 256, 256, 0, stream>>>(
            score, dst, nmax, wexp, nsum, E);

        gatv2_edge_aggregate<<<(E + 7) / 8, 256, 0, stream>>>(
            xl, wexp, nsum, src, dst, acc, E, O);

        gatv2_finalize<<<(NO + 255) / 256, 256, 0, stream>>>(
            acc, L[li].b, outs[li], NO, O, li == 2 ? 1 : 0);

        xin = outs[li];
    }
}
